// MambaBlock_48069273977389
// MI455X (gfx1250) — compile-verified, matrix operand order checked by reading, not run on hardware
//
#include <hip/hip_runtime.h>
#include <hip/hip_bf16.h>
#include <math.h>

// ---------------------------------------------------------------------------
// Mamba block forward for MI455X (gfx1250):
//   bf16 WMMA GEMMs with async-DMA (GLOBAL_LOAD_ASYNC_TO_LDS_B128) staging,
//   double-buffered LDS, parallel selective scan.
// ---------------------------------------------------------------------------

#define D_MODEL 768
#define D_INNER 1536
#define D_STATE 16
#define D_CONV  4
#define DT_RANK 48
#define XDBL_N  (DT_RANK + 2 * D_STATE)   // 80

typedef __attribute__((ext_vector_type(16))) __bf16 v16bf;
typedef __attribute__((ext_vector_type(8)))  __bf16 v8bf;
typedef __attribute__((ext_vector_type(8)))  float  v8f;

// LDS row stride (bf16 elems): 32 data + 8 pad -> 80B row stride, 16B aligned,
// skews banks (80*i mod 256 distinct for i=0..15) -> conflict-free b128 reads.
#define LROW 40

__device__ __forceinline__ float silu_f(float x) {
    return x / (1.f + __expf(-x));
}

// 16B global -> LDS async DMA (ASYNCcnt-tracked, no VGPR round trip).
__device__ __forceinline__ void async_cp16(unsigned int lds_off,
                                           const void* gptr) {
    asm volatile("global_load_async_to_lds_b128 %0, %1, off"
                 :: "v"(lds_off), "v"((unsigned long long)(uintptr_t)gptr)
                 : "memory");
}
__device__ __forceinline__ void wait_async0() {
    asm volatile("s_wait_asynccnt 0x0" ::: "memory");
}
__device__ __forceinline__ unsigned int lds_off_of(const void* p) {
    return (unsigned int)(uintptr_t)p;   // generic LDS ptr: low 32b = LDS addr
}

// ISA 16-bit fragment layout: lanes 0-15 hold K {kb..kb+7, kb+16..kb+23} with
// kb = 0; lanes 16-31 with kb = 8.  rowBase points at a 32-K bf16 LDS row.
__device__ __forceinline__ v16bf ld_frag(const __bf16* rowBase, int half) {
    union { v16bf v; v8bf h[2]; } u;
    const v8bf* p = (const v8bf*)rowBase;
    u.h[0] = p[half];
    u.h[1] = p[half + 2];
    return u.v;
}

#define WMMA_BF16(A_, B_, C_) \
    __builtin_amdgcn_wmma_f32_16x16x32_bf16(false, (A_), false, (B_), (short)0, (C_), false, false)

// ---------------------------------------------------------------------------
// fp32 -> bf16 pre-conversion (weights / input activations), float4 granular.
// ---------------------------------------------------------------------------
__global__ void cvt_f32_bf16(const float* __restrict__ s,
                             __bf16* __restrict__ d, int n4) {
    const int i = blockIdx.x * blockDim.x + threadIdx.x;
    if (i < n4) {
        const float4 f = ((const float4*)s)[i];
        __bf16* p = d + (size_t)i * 4;
        p[0] = (__bf16)f.x; p[1] = (__bf16)f.y;
        p[2] = (__bf16)f.z; p[3] = (__bf16)f.w;
    }
}

// ---------------------------------------------------------------------------
// Big GEMM: C[M,N] = epi( A[M,K]bf16 (row stride lda) @ W[N,K]bf16^T ), f32 C.
// Requires M % 64 == 0, N % 128 == 0.  Block 256 thr = 8 waves as 2(M)x4(N);
// block tile 64x128; each wave computes a 32x32 quad (4 WMMAs / K-step).
// Async DMA staging, double-buffered LDS: tile t+1 DMA overlaps tile t WMMAs.
// EPI: 0 = none; 1 = softplus(x + bias[n]).  KTAIL: handle K % 32 != 0 once.
// ---------------------------------------------------------------------------
template <int EPI, bool KTAIL>
__global__ void __launch_bounds__(256)
gemm_big(const __bf16* __restrict__ A, int lda,
         const __bf16* __restrict__ W,
         const float* __restrict__ bias,
         float* __restrict__ C,
         int M, int N, int K) {
    __shared__ __align__(16) __bf16 lsA[2][64 * LROW];
    __shared__ __align__(16) __bf16 lsB[2][128 * LROW];

    const int tid  = threadIdx.x;
    const int lane = tid & 31;
    const int wave = tid >> 5;          // 0..7
    const int wm   = wave >> 2;         // 0..1
    const int wn   = wave & 3;          // 0..3
    const int ml   = lane & 15;
    const int half = lane >> 4;

    const int mBase = blockIdx.y * 64;
    const int nBase = blockIdx.x * 128;

    // 16B chunk mapping: 4 chunks per 32-K bf16 row.
    const int cr = tid >> 2;            // 0..63
    const int cg = tid & 3;             // chunk-in-row

    // Global sources for this thread's chunks (advance 32 elems per K-step).
    const __bf16* gA  = A + (size_t)(mBase + cr) * lda + cg * 8;
    const __bf16* gB0 = W + (size_t)(nBase + cr     ) * K + cg * 8;
    const __bf16* gB1 = W + (size_t)(nBase + cr + 64) * K + cg * 8;

    // Fixed LDS destinations per buffer.
    unsigned int dA[2], dB0[2], dB1[2];
    #pragma unroll
    for (int b = 0; b < 2; ++b) {
        dA[b]  = lds_off_of(&lsA[b][cr * LROW + cg * 8]);
        dB0[b] = lds_off_of(&lsB[b][cr * LROW + cg * 8]);
        dB1[b] = lds_off_of(&lsB[b][(cr + 64) * LROW + cg * 8]);
    }

    // Fragment row bases per buffer.
    const __bf16* fra[2][2]; const __bf16* frb[2][2];
    #pragma unroll
    for (int b = 0; b < 2; ++b) {
        fra[b][0] = &lsA[b][(wm * 32      + ml) * LROW];
        fra[b][1] = &lsA[b][(wm * 32 + 16 + ml) * LROW];
        frb[b][0] = &lsB[b][(wn * 32      + ml) * LROW];
        frb[b][1] = &lsB[b][(wn * 32 + 16 + ml) * LROW];
    }

    const v8f vzero = {0.f, 0.f, 0.f, 0.f, 0.f, 0.f, 0.f, 0.f};
    v8f acc00 = vzero, acc01 = vzero, acc10 = vzero, acc11 = vzero;

    const int FS = K >> 5;              // full 32-K steps
    auto issue = [&](int kt, int buf) {
        async_cp16(dA[buf],  gA  + kt * 32);
        async_cp16(dB0[buf], gB0 + kt * 32);
        async_cp16(dB1[buf], gB1 + kt * 32);
    };

    if (FS > 0) issue(0, 0);
    for (int kt = 0; kt < FS; ++kt) {
        const int cur = kt & 1;
        wait_async0();                  // this wave's DMA into buf[cur] done
        __syncthreads();                // all waves' DMA visible; prev reads done
        if (kt + 1 < FS) issue(kt + 1, cur ^ 1);   // DMA under the WMMAs
        const v16bf fa0 = ld_frag(fra[cur][0], half);
        const v16bf fa1 = ld_frag(fra[cur][1], half);
        const v16bf fb0 = ld_frag(frb[cur][0], half);
        const v16bf fb1 = ld_frag(frb[cur][1], half);
        acc00 = WMMA_BF16(fa0, fb0, acc00);
        acc01 = WMMA_BF16(fa0, fb1, acc01);
        acc10 = WMMA_BF16(fa1, fb0, acc10);
        acc11 = WMMA_BF16(fa1, fb1, acc11);
    }

    if (KTAIL) {                        // single zero-padded tail step
        const int k0  = FS << 5;
        const int buf = FS & 1;
        #pragma unroll
        for (int j = 0; j < 8; ++j) {
            const int kk = k0 + cg * 8 + j;
            const bool ok = kk < K;
            lsA[buf][cr * LROW + cg * 8 + j] =
                ok ? A[(size_t)(mBase + cr) * lda + kk] : (__bf16)0.f;
            lsB[buf][cr * LROW + cg * 8 + j] =
                ok ? W[(size_t)(nBase + cr) * K + kk] : (__bf16)0.f;
            lsB[buf][(cr + 64) * LROW + cg * 8 + j] =
                ok ? W[(size_t)(nBase + cr + 64) * K + kk] : (__bf16)0.f;
        }
        __syncthreads();
        const v16bf fa0 = ld_frag(fra[buf][0], half);
        const v16bf fa1 = ld_frag(fra[buf][1], half);
        const v16bf fb0 = ld_frag(frb[buf][0], half);
        const v16bf fb1 = ld_frag(frb[buf][1], half);
        acc00 = WMMA_BF16(fa0, fb0, acc00);
        acc01 = WMMA_BF16(fa0, fb1, acc01);
        acc10 = WMMA_BF16(fa1, fb0, acc10);
        acc11 = WMMA_BF16(fa1, fb1, acc11);
    }

    // Epilogue.  C layout: lanes 0-15 M=r, lanes 16-31 M=8+r; N = lane&15.
    auto storeTile = [&](const v8f& a, int mi, int ni) {
        const int gn = nBase + wn * 32 + ni * 16 + ml;
        #pragma unroll
        for (int r = 0; r < 8; ++r) {
            const int gm = mBase + wm * 32 + mi * 16 + half * 8 + r;
            float v = a[r];
            if (EPI == 1) {
                v += bias[gn];
                v = (v > 20.f) ? v : log1pf(__expf(v));   // softplus
            }
            C[(size_t)gm * N + gn] = v;
        }
    };
    storeTile(acc00, 0, 0); storeTile(acc01, 0, 1);
    storeTile(acc10, 1, 0); storeTile(acc11, 1, 1);
}

// ---------------------------------------------------------------------------
// x_proj GEMM: xdbl[M,80] = u_bf[M,K] @ Wbf[80,K]^T (all bf16 in, f32 out).
// Also emits bf16 copy of the first 48 columns (dt input, stride 48).
// Requires M % 128 == 0, K % 32 == 0.  8 waves, each owns 16 rows x 5 N-tiles.
// ---------------------------------------------------------------------------
__global__ void __launch_bounds__(256)
gemm_n80(const __bf16* __restrict__ A,
         const __bf16* __restrict__ W,
         float* __restrict__ C,
         __bf16* __restrict__ C48bf,
         int M, int K) {
    __shared__ __align__(16) __bf16 lsA[128 * LROW];
    __shared__ __align__(16) __bf16 lsB[XDBL_N * LROW];

    const int tid  = threadIdx.x;
    const int lane = tid & 31;
    const int wave = tid >> 5;
    const int ml   = lane & 15;
    const int half = lane >> 4;
    const int mBase = blockIdx.x * 128;
    const int cr = tid >> 2, cg = tid & 3;

    const v8f vzero = {0.f, 0.f, 0.f, 0.f, 0.f, 0.f, 0.f, 0.f};
    v8f acc[5];
    #pragma unroll
    for (int nt = 0; nt < 5; ++nt) acc[nt] = vzero;

    const int FS = K >> 5;
    for (int kt = 0; kt < FS; ++kt) {
        const int k0 = kt << 5;
        __syncthreads();
        // A: 128 rows * 4 chunks = 512, 2 per thread (pure 16B copies).
        #pragma unroll
        for (int j = 0; j < 2; ++j) {
            const int row = cr + j * 64;
            *(v8bf*)&lsA[row * LROW + cg * 8] =
                *(const v8bf*)(A + (size_t)(mBase + row) * K + k0 + cg * 8);
        }
        // B: 80 rows * 4 chunks = 320.
        for (int i = tid; i < XDBL_N * 4; i += 256) {
            const int r = i >> 2, g = i & 3;
            *(v8bf*)&lsB[r * LROW + g * 8] =
                *(const v8bf*)(W + (size_t)r * K + k0 + g * 8);
        }
        __syncthreads();
        const v16bf fa = ld_frag(&lsA[(wave * 16 + ml) * LROW], half);
        #pragma unroll
        for (int nt = 0; nt < 5; ++nt) {
            const v16bf fb = ld_frag(&lsB[(nt * 16 + ml) * LROW], half);
            acc[nt] = WMMA_BF16(fa, fb, acc[nt]);
        }
    }
    #pragma unroll
    for (int nt = 0; nt < 5; ++nt) {
        const int gn = nt * 16 + ml;
        #pragma unroll
        for (int r = 0; r < 8; ++r) {
            const int gm = mBase + wave * 16 + half * 8 + r;
            const float v = acc[nt][r];
            C[(size_t)gm * XDBL_N + gn] = v;
            if (nt < 3)                      // first 48 cols -> bf16 dt input
                C48bf[(size_t)gm * DT_RANK + gn] = (__bf16)v;
        }
    }
}

// ---------------------------------------------------------------------------
// Causal depthwise conv (width 4) + bias + SiLU -> u (f32 for scan) and
// u_bf (bf16 for the x_proj GEMM).
// ---------------------------------------------------------------------------
__global__ void mamba_conv_silu(const float* __restrict__ xz,
                                const float* __restrict__ cw,
                                const float* __restrict__ cb,
                                float* __restrict__ u,
                                __bf16* __restrict__ ub,
                                int total, int L) {
    const int idx = blockIdx.x * blockDim.x + threadIdx.x;
    if (idx >= total) return;
    const int d = idx % D_INNER;
    const int m = idx / D_INNER;      // b*L + l
    const int l = m % L;
    float acc = cb[d];
    #pragma unroll
    for (int j = 0; j < D_CONV; ++j) {
        const int lt = l - (D_CONV - 1) + j;
        if (lt >= 0)
            acc += cw[d * D_CONV + j] * xz[(size_t)(m - (D_CONV - 1) + j) * (2 * D_INNER) + d];
    }
    const float v = silu_f(acc);
    u[idx]  = v;
    ub[idx] = (__bf16)v;
}

// ---------------------------------------------------------------------------
// Selective scan. Lane = state s (16 lanes per channel), 2 channels per wave.
// Sequential over t; y_bf[m,d] = (sum_s h*C + D*u) * silu(z)  (bf16: feeds
// only the out_proj WMMA GEMM).
// ---------------------------------------------------------------------------
__global__ void __launch_bounds__(256)
mamba_scan(const float* __restrict__ xz,
           const float* __restrict__ u,
           const float* __restrict__ xdbl,
           const float* __restrict__ delta,
           const float* __restrict__ A_log,
           const float* __restrict__ Dp,
           __bf16* __restrict__ yb,
           int L) {
    const int lane = threadIdx.x & 31;
    const int wave = (blockIdx.x * blockDim.x + threadIdx.x) >> 5;
    const int ch   = wave * 2 + (lane >> 4);        // b*D_INNER + d
    const int b = ch / D_INNER, d = ch % D_INNER;
    const int s = lane & 15;

    const float As = -__expf(A_log[d * D_STATE + s]);
    const float Dd = Dp[d];
    float h = 0.f;

    const size_t base80 = (size_t)b * L * XDBL_N;
    const size_t baseI  = (size_t)b * L * D_INNER;
    const size_t baseZ  = (size_t)b * L * (2 * D_INNER);

    for (int t = 0; t < L; ++t) {
        const size_t ri = baseI + (size_t)t * D_INNER + d;
        const float du = delta[ri];
        const float uu = u[ri];
        const size_t r80 = base80 + (size_t)t * XDBL_N;
        const float bs = xdbl[r80 + DT_RANK + s];
        const float cs = xdbl[r80 + DT_RANK + D_STATE + s];

        h = __expf(du * As) * h + (du * uu) * bs;
        float p = h * cs;
        p += __shfl_xor(p, 1, 16);
        p += __shfl_xor(p, 2, 16);
        p += __shfl_xor(p, 4, 16);
        p += __shfl_xor(p, 8, 16);
        if (s == 0) {
            const float zz = xz[baseZ + (size_t)t * (2 * D_INNER) + D_INNER + d];
            yb[ri] = (__bf16)((p + Dd * uu) * silu_f(zz));
        }
    }
}

// ---------------------------------------------------------------------------
// LayerNorm over last dim (768) + residual. One 256-thread block per row.
// ---------------------------------------------------------------------------
__global__ void __launch_bounds__(256)
mamba_ln_res(const float* __restrict__ v_in,
             const float* __restrict__ res,
             const float* __restrict__ w,
             const float* __restrict__ bsh,
             float* __restrict__ outF) {
    const int row = blockIdx.x;
    const float* v = v_in + (size_t)row * D_MODEL;

    float s = 0.f, s2 = 0.f;
    for (int i = threadIdx.x; i < D_MODEL; i += 256) {
        const float t = v[i];
        s += t; s2 += t * t;
    }
    for (int m = 1; m < 32; m <<= 1) {
        s  += __shfl_xor(s,  m, 32);
        s2 += __shfl_xor(s2, m, 32);
    }
    __shared__ float as[8], as2[8];
    const int wv = threadIdx.x >> 5, ln = threadIdx.x & 31;
    if (ln == 0) { as[wv] = s; as2[wv] = s2; }
    __syncthreads();
    if (wv == 0) {
        s  = (ln < 8) ? as[ln]  : 0.f;
        s2 = (ln < 8) ? as2[ln] : 0.f;
        for (int m = 1; m < 8; m <<= 1) {
            s  += __shfl_xor(s,  m, 32);
            s2 += __shfl_xor(s2, m, 32);
        }
        if (ln == 0) { as[0] = s; as2[0] = s2; }
    }
    __syncthreads();
    const float mu  = as[0] * (1.f / D_MODEL);
    const float var = as2[0] * (1.f / D_MODEL) - mu * mu;
    const float inv = rsqrtf(var + 1e-5f);
    for (int i = threadIdx.x; i < D_MODEL; i += 256) {
        outF[(size_t)row * D_MODEL + i] =
            (v[i] - mu) * inv * w[i] + bsh[i] + res[(size_t)row * D_MODEL + i];
    }
}

// ---------------------------------------------------------------------------
extern "C" void kernel_launch(void* const* d_in, const int* in_sizes, int n_in,
                              void* d_out, int out_size, void* d_ws, size_t ws_size,
                              hipStream_t stream) {
    const float* x         = (const float*)d_in[0];
    const float* in_proj_w = (const float*)d_in[1];
    const float* conv_w    = (const float*)d_in[2];
    const float* conv_b    = (const float*)d_in[3];
    const float* x_proj_w  = (const float*)d_in[4];
    const float* dt_proj_w = (const float*)d_in[5];
    const float* dt_proj_b = (const float*)d_in[6];
    const float* A_log     = (const float*)d_in[7];
    const float* Dp        = (const float*)d_in[8];
    const float* out_proj_w= (const float*)d_in[9];
    const float* ln_w      = (const float*)d_in[10];
    const float* ln_b      = (const float*)d_in[11];

    const int M = in_sizes[0] / D_MODEL;   // B*L = 4096
    const int L = 2048;
    const int B = M / L;

    // ---- workspace layout: fp32 region, then bf16 region (16B aligned) ----
    float* ws    = (float*)d_ws;
    float* xz    = ws;                                  // M * 3072
    float* u     = xz    + (size_t)M * (2 * D_INNER);   // M * 1536
    float* xdbl  = u     + (size_t)M * D_INNER;         // M * 80
    float* delta = xdbl  + (size_t)M * XDBL_N;          // M * 1536
    float* outp  = delta + (size_t)M * D_INNER;         // M * 768

    __bf16* bfw   = (__bf16*)(outp + (size_t)M * D_MODEL);
    __bf16* x_bf  = bfw;                                       // M*768
    __bf16* u_bf  = x_bf  + (size_t)M * D_MODEL;               // M*1536
    __bf16* y_bf  = u_bf  + (size_t)M * D_INNER;               // M*1536
    __bf16* dt_bf = y_bf  + (size_t)M * D_INNER;               // M*48
    __bf16* w_in  = dt_bf + (size_t)M * DT_RANK;               // 3072*768
    __bf16* w_xp  = w_in  + (size_t)(2 * D_INNER) * D_MODEL;   // 80*1536
    __bf16* w_dt  = w_xp  + (size_t)XDBL_N * D_INNER;          // 1536*48
    __bf16* w_out = w_dt  + (size_t)D_INNER * DT_RANK;         // 768*1536

    auto cdiv = [](int a, int b) { return (a + b - 1) / b; };
    const dim3 blk(256);

    // 0) pre-convert input + weights to bf16 (once per launch, deterministic)
    const int nx  = M * D_MODEL / 4;
    const int nwi = (2 * D_INNER) * D_MODEL / 4;
    const int nwx = XDBL_N * D_INNER / 4;
    const int nwd = D_INNER * DT_RANK / 4;
    const int nwo = D_MODEL * D_INNER / 4;
    cvt_f32_bf16<<<cdiv(nx,  256), blk, 0, stream>>>(x,          x_bf, nx);
    cvt_f32_bf16<<<cdiv(nwi, 256), blk, 0, stream>>>(in_proj_w,  w_in, nwi);
    cvt_f32_bf16<<<cdiv(nwx, 256), blk, 0, stream>>>(x_proj_w,   w_xp, nwx);
    cvt_f32_bf16<<<cdiv(nwd, 256), blk, 0, stream>>>(dt_proj_w,  w_dt, nwd);
    cvt_f32_bf16<<<cdiv(nwo, 256), blk, 0, stream>>>(out_proj_w, w_out, nwo);

    // 1) in_proj: xz[M,3072] = x @ in_proj_w^T
    gemm_big<0, false><<<dim3((2 * D_INNER) / 128, M / 64), blk, 0, stream>>>(
        x_bf, D_MODEL, w_in, nullptr, xz, M, 2 * D_INNER, D_MODEL);

    // 2) causal depthwise conv + SiLU -> u (f32) + u_bf (bf16)
    mamba_conv_silu<<<cdiv(M * D_INNER, 256), blk, 0, stream>>>(
        xz, conv_w, conv_b, u, u_bf, M * D_INNER, L);

    // 3) x_proj: xdbl[M,80] = u @ x_proj_w^T  (+ bf16 copy of first 48 cols)
    gemm_n80<<<M / 128, blk, 0, stream>>>(u_bf, w_xp, xdbl, dt_bf, M, D_INNER);

    // 4) dt_proj + softplus: delta = softplus(xdbl[:, :48] @ dt_proj_w^T + b)
    gemm_big<1, true><<<dim3(D_INNER / 128, M / 64), blk, 0, stream>>>(
        dt_bf, DT_RANK, w_dt, dt_proj_b, delta, M, D_INNER, DT_RANK);

    // 5) selective scan -> y_bf (includes D*u skip and silu(z) gate)
    const int channels = B * D_INNER;                 // 3072
    mamba_scan<<<cdiv(channels, 16), blk, 0, stream>>>(
        xz, u, xdbl, delta, A_log, Dp, y_bf, L);

    // 6) out_proj: outp[M,768] = y @ out_proj_w^T
    gemm_big<0, false><<<dim3(D_MODEL / 128, M / 64), blk, 0, stream>>>(
        y_bf, D_INNER, w_out, nullptr, outp, M, D_MODEL, D_INNER);

    // 7) LayerNorm + residual -> d_out
    mamba_ln_res<<<M, blk, 0, stream>>>(outp, x, ln_w, ln_b, (float*)d_out);
}
